// KappaGCNLayer_29240137351291
// MI455X (gfx1250) — matrix operand order repaired and checked by reading, not compiled, not run on hardware
//
#include <hip/hip_runtime.h>
#include <hip/hip_bf16.h>

// ---------------------------------------------------------------------------
// KappaGCN layer (Poincare ball, kappa = -1) for MI455X / gfx1250.
// Dominant cost: A_hat(8192x8192) @ Y(8192x528) GEMM -> bf16 WMMA, f32 accum.
// LDS tiles are stored in fragment-native order so every WMMA operand is
// loaded with two ds_load_b128 (no scalar ds_load_u16 gather).
// ---------------------------------------------------------------------------

#define NROWS 8192
#define DDIM  512
#define LDY   528        // 512 features + col512=(gamma-1), col513=1.0, pad to 33*16
#define EPSF  1e-10f

typedef __attribute__((ext_vector_type(16))) __bf16 v16bf;
typedef __attribute__((ext_vector_type(8)))  float  v8f;

union Frag16 { v16bf v; uint4 q[2]; unsigned short u[16]; };

// round-to-nearest-even f32 -> bf16 bit pattern
__device__ __forceinline__ unsigned short f2bfu(float f) {
    union { float f; unsigned u; } x; x.f = f;
    unsigned r = x.u + 0x7FFFu + ((x.u >> 16) & 1u);
    return (unsigned short)(r >> 16);
}
// pack two floats into (lo,hi) bf16 pair
__device__ __forceinline__ unsigned f2bfu2(float a, float b) {
    return (unsigned)f2bfu(a) | ((unsigned)f2bfu(b) << 16);
}

__device__ __forceinline__ float artanh_c(float x) {
    float z = fminf(fmaxf(x, -0.9999999f), 0.9999999f);
    return 0.5f * __logf((1.f + z) / (1.f - z));
}

__device__ __forceinline__ float waveSum(float x) {
    // wave32 butterfly reduction
    #pragma unroll
    for (int off = 16; off > 0; off >>= 1)
        x += __shfl_xor(x, off, 32);
    return x;
}

// ---------------------------------------------------------------------------
// k0: convert W to bf16
// ---------------------------------------------------------------------------
__global__ __launch_bounds__(256) void k0_convW(const float* __restrict__ W,
                                                unsigned short* __restrict__ Wb,
                                                int n) {
    int i = blockIdx.x * 256 + threadIdx.x;
    if (i < n) Wb[i] = f2bfu(W[i]);
}

// ---------------------------------------------------------------------------
// GEMM: C[M,N] = A[M,K](f32 -> bf16 in flight) @ B[K,N](bf16)
// Workgroup tile 128(M) x 128(N), K-step 32; 8 waves as 4(M) x 2(N);
// each wave computes 32x64 = 8 x v_wmma_f32_16x16x32_bf16 per K-step.
// LDS:  lA  [128][40]  row-major  (A fragment = 2 aligned uint4 per lane)
//       lBt [128][40]  TRANSPOSED (B fragment = 2 aligned uint4 per lane)
// Requires: M % 128 == 0, K % 32 == 0, N even. N guarded.
// ---------------------------------------------------------------------------
#define TM 128
#define TN 128
#define TK 32
#define LSTR 40   // ushorts per LDS row: 80 bytes, 16B-aligned, 20 dwords

__global__ __launch_bounds__(256) void gemm_bf16(const float* __restrict__ A,
                                                 const unsigned short* __restrict__ B,
                                                 float* __restrict__ C,
                                                 int M, int N, int K,
                                                 int lda, int ldb, int ldc) {
    __shared__ __align__(16) unsigned short lA[TM][LSTR];
    __shared__ __align__(16) unsigned short lBt[TN][LSTR];

    const int m0 = blockIdx.y * TM;
    const int n0 = blockIdx.x * TN;
    const int tid  = threadIdx.x;
    const int lane = tid & 31;
    const int wave = tid >> 5;       // 0..7
    const int wm = wave >> 1;        // 0..3: 32-row block within tile
    const int wn = wave & 1;         // 0..1: 64-col block within tile

    v8f acc[2][4] = {};              // [m-sub 16][n-sub 16]

    const int lan = lane & 15;
    const int hi  = (lane & 16) ? 1 : 0;   // second half-lane group

    for (int k0 = 0; k0 < K; k0 += TK) {
        // ---- stage A tile: 128x32 f32 -> bf16, float2 loads, b32 LDS stores
        #pragma unroll
        for (int i = 0; i < 8; ++i) {
            int idx = tid + i * 256;             // 0..2047 float2 slots
            int r = idx >> 4;                    // 0..127
            int c = (idx & 15) << 1;             // 0,2,..,30
            float2 v = *(const float2*)&A[(long)(m0 + r) * lda + k0 + c];
            *(unsigned*)&lA[r][c] = f2bfu2(v.x, v.y);
        }
        // prefetch hint for next A tile (emits global_prefetch_b8)
        if (k0 + TK < K)
            __builtin_prefetch(&A[(long)(m0 + (tid >> 1)) * lda + k0 + TK], 0, 1);
        // ---- stage B tile transposed: read 2x2 bf16 blocks, store packed ----
        #pragma unroll
        for (int i = 0; i < 4; ++i) {
            int idx = tid + i * 256;             // 0..1023 (2x2 blocks)
            int k2 = (idx >> 6) << 1;            // 0,2,..,30
            int n2 = (idx & 63) << 1;            // 0,2,..,126
            int col = n0 + n2;
            unsigned u0 = 0, u1 = 0;
            if (col < N) {
                u0 = *(const unsigned*)&B[(long)(k0 + k2) * ldb + col];
                u1 = *(const unsigned*)&B[(long)(k0 + k2 + 1) * ldb + col];
            }
            *(unsigned*)&lBt[n2][k2]     = (u0 & 0xFFFFu) | (u1 << 16);
            *(unsigned*)&lBt[n2 + 1][k2] = (u0 >> 16) | (u1 & 0xFFFF0000u);
        }
        __syncthreads();

        // ---- load fragments: every operand = 2 aligned ds_load_b128 ----
        Frag16 af[2];
        #pragma unroll
        for (int ms = 0; ms < 2; ++ms) {
            const uint4* rowp =
                (const uint4*)&lA[wm * 32 + ms * 16 + lan][0];
            af[ms].q[0] = rowp[hi];          // K = 8*hi      .. +7
            af[ms].q[1] = rowp[hi + 2];      // K = 16 + 8*hi .. +7
        }
        Frag16 bf_[4];
        #pragma unroll
        for (int t = 0; t < 4; ++t) {
            const uint4* colp =
                (const uint4*)&lBt[wn * 64 + t * 16 + lan][0];
            bf_[t].q[0] = colp[hi];          // K = 8*hi      .. +7
            bf_[t].q[1] = colp[hi + 2];      // K = 16 + 8*hi .. +7
        }
        // ---- 8 WMMAs per wave per K-step ----
        #pragma unroll
        for (int ms = 0; ms < 2; ++ms)
            #pragma unroll
            for (int t = 0; t < 4; ++t)
                acc[ms][t] = __builtin_amdgcn_wmma_f32_16x16x32_bf16(
                    false, af[ms].v, false, bf_[t].v, (short)0, acc[ms][t],
                    false, false);
        __syncthreads();
    }

    // ---- store C per CDNA5 16x16 f32 C/D layout ----
    #pragma unroll
    for (int ms = 0; ms < 2; ++ms) {
        int mrow0 = m0 + wm * 32 + ms * 16 + hi * 8;   // lanes 16-31 -> M+8
        #pragma unroll
        for (int t = 0; t < 4; ++t) {
            int col = n0 + wn * 64 + t * 16 + lan;
            if (col < N) {
                #pragma unroll
                for (int r = 0; r < 8; ++r)
                    C[(long)(mrow0 + r) * ldc + col] = acc[ms][t][r];
            }
        }
    }
}

// ---------------------------------------------------------------------------
// k2: per-row Mobius-matvec epilogue + gamma; one wave per row (16 elems/lane)
//   res  = tanh((|mx|/|x|) * artanh(|x|)) * mx/|mx|   (0 if |mx|<=eps)
//   gamma= max(2/(1 - |res|^2), 1e-15)
//   Y[:, 0:512] = bf16(gamma*res); Y[:,512]=bf16(gamma-1); Y[:,513]=1; pad 0
// ---------------------------------------------------------------------------
__global__ __launch_bounds__(256) void k2_mobius_gamma(const float* __restrict__ X,
                                                       const float* __restrict__ MX,
                                                       unsigned short* __restrict__ Y) {
    int row  = blockIdx.x * 8 + (threadIdx.x >> 5);
    int lane = threadIdx.x & 31;
    const float* xr = X  + (long)row * DDIM;
    const float* mr = MX + (long)row * DDIM;

    float mx[16];
    float xs = 0.f, ms = 0.f;
    #pragma unroll
    for (int i = 0; i < 16; ++i) {
        float xv = xr[lane + i * 32];
        float mv = mr[lane + i * 32];
        mx[i] = mv;
        xs += xv * xv;
        ms += mv * mv;
    }
    xs = waveSum(xs); ms = waveSum(ms);
    float xn = sqrtf(fmaxf(xs, EPSF * EPSF));
    float mn = sqrtf(fmaxf(ms, EPSF * EPSF));
    float scale = tanhf(mn / xn * artanh_c(xn)) / mn;
    bool  zero  = (mn <= EPSF);

    float res[16];
    float rs2 = 0.f;
    #pragma unroll
    for (int i = 0; i < 16; ++i) {
        res[i] = zero ? 0.f : scale * mx[i];
        rs2 += res[i] * res[i];
    }
    rs2 = waveSum(rs2);
    float gamma = fmaxf(2.f / (1.f - rs2), 1e-15f);   // kappa = -1

    unsigned short* yr = Y + (long)row * LDY;
    #pragma unroll
    for (int i = 0; i < 16; ++i)
        yr[lane + i * 32] = f2bfu(gamma * res[i]);
    if (lane == 0) {
        yr[DDIM]     = f2bfu(gamma - 1.f);
        yr[DDIM + 1] = f2bfu(1.f);
        for (int c = DDIM + 2; c < LDY; ++c) yr[c] = 0;
    }
}

// ---------------------------------------------------------------------------
// k4: per-row final chain; one wave per row.
//   nom=G[:,0:512]; denom=clamp_abs(G[:,512]); alpha=G[:,513]
//   two_mean=nom/denom; a_mean=two_mean/(1+sqrt(max(1-|two_mean|^2,eps)))
//   ms = tanh(alpha*artanh(|a_mean|)) * a_mean/|a_mean|
//   u  = relu(artanh(|ms|) * ms/|ms|) ; out = tanh(|u|) * u/|u|
// ---------------------------------------------------------------------------
__global__ __launch_bounds__(256) void k4_final(const float* __restrict__ G,
                                                float* __restrict__ out) {
    int row  = blockIdx.x * 8 + (threadIdx.x >> 5);
    int lane = threadIdx.x & 31;
    const float* gr = G + (long)row * LDY;

    float denomRaw = gr[DDIM];
    float alpha    = gr[DDIM + 1];
    float denom = ((denomRaw >= 0.f) ? 1.f : -1.f) * fmaxf(fabsf(denomRaw), EPSF);

    float tm[16];
    float sq = 0.f;
    #pragma unroll
    for (int i = 0; i < 16; ++i) {
        tm[i] = gr[lane + i * 32] / denom;
        sq += tm[i] * tm[i];
    }
    sq = waveSum(sq);
    float aden = 1.f + sqrtf(fmaxf(1.f - sq, EPSF));   // 1 + sqrt(1 + kappa*sq)

    float am[16];
    float s2 = 0.f;
    #pragma unroll
    for (int i = 0; i < 16; ++i) { am[i] = tm[i] / aden; s2 += am[i] * am[i]; }
    s2 = waveSum(s2);
    float an  = sqrtf(fmaxf(s2, EPSF * EPSF));
    float msc = tanhf(alpha * artanh_c(an)) / an;      // mobius_scalar_mul

    float msv[16];
    float s3 = 0.f;
    #pragma unroll
    for (int i = 0; i < 16; ++i) { msv[i] = msc * am[i]; s3 += msv[i] * msv[i]; }
    s3 = waveSum(s3);
    float msn = sqrtf(fmaxf(s3, EPSF * EPSF));
    float lsc = artanh_c(msn) / msn;                   // logmap0

    float u[16];
    float s4 = 0.f;
    #pragma unroll
    for (int i = 0; i < 16; ++i) { u[i] = fmaxf(lsc * msv[i], 0.f); s4 += u[i] * u[i]; }
    s4 = waveSum(s4);
    float un  = sqrtf(fmaxf(s4, EPSF * EPSF));
    float esc = tanhf(un) / un;                        // expmap0

    float* orow = out + (long)row * DDIM;
    #pragma unroll
    for (int i = 0; i < 16; ++i)
        orow[lane + i * 32] = esc * u[i];
}

// ---------------------------------------------------------------------------
extern "C" void kernel_launch(void* const* d_in, const int* in_sizes, int n_in,
                              void* d_out, int out_size, void* d_ws, size_t ws_size,
                              hipStream_t stream) {
    const float* X = (const float*)d_in[0];   // [8192, 512]
    const float* A = (const float*)d_in[1];   // [8192, 8192]
    const float* W = (const float*)d_in[2];   // [512, 512]
    float* out = (float*)d_out;               // [8192, 512]

    char* ws = (char*)d_ws;
    size_t off = 0;
    unsigned short* Wb = (unsigned short*)(ws + off);          // 512*512*2 = 0.5 MB
    off += (size_t)DDIM * DDIM * 2;      off = (off + 255) & ~(size_t)255;
    float* MX = (float*)(ws + off);                            // 8192*512*4 = 16 MB
    off += (size_t)NROWS * DDIM * 4;     off = (off + 255) & ~(size_t)255;
    unsigned short* Y = (unsigned short*)(ws + off);           // 8192*528*2 = 8.25 MB
    off += (size_t)NROWS * LDY * 2;      off = (off + 255) & ~(size_t)255;
    float* G = (float*)(ws + off);                             // 8192*528*4 = 16.5 MB

    // k0: W -> bf16
    k0_convW<<<(DDIM * DDIM + 255) / 256, 256, 0, stream>>>(W, Wb, DDIM * DDIM);

    // k1: MX = X @ W   (8192 x 512 x 512)
    dim3 g1(DDIM / TN, NROWS / TM);
    gemm_bf16<<<g1, 256, 0, stream>>>(X, Wb, MX, NROWS, DDIM, DDIM,
                                      DDIM, DDIM, DDIM);

    // k2: Mobius rescale + gamma, build Y (bf16, 528 cols)
    k2_mobius_gamma<<<NROWS / 8, 256, 0, stream>>>(X, MX, Y);

    // k3: G = A_hat @ Y  (8192 x 528 x 8192)  -- dominant GEMM
    dim3 g2((LDY + TN - 1) / TN, NROWS / TM);
    gemm_bf16<<<g2, 256, 0, stream>>>(A, Y, G, NROWS, LDY, NROWS,
                                      NROWS, LDY, LDY);

    // k4: midpoint + scalar-mul + log0/relu/exp0 -> out
    k4_final<<<NROWS / 8, 256, 0, stream>>>(G, out);
}
